// AutoregressiveDecoder_25116968747341
// MI455X (gfx1250) — matrix operand order repaired and checked by reading, not compiled, not run on hardware
//
#include <hip/hip_runtime.h>
#include <hip/hip_bf16.h>

#define N_      384
#define IN_DIM_ 128
#define HID_    256
#define HID2_   128

typedef unsigned short u16;
typedef __attribute__((ext_vector_type(16))) __bf16        bf16x16;
typedef __attribute__((ext_vector_type(8)))  float         v8f;
typedef __attribute__((ext_vector_type(4)))  unsigned int  u32x4;
typedef __attribute__((ext_vector_type(4)))  float         f32x4;

// round-half-up f32->bf16; pack two into one dword with a single v_perm_b32
__device__ __forceinline__ unsigned int pk2(float a, float b) {
  unsigned int ua = __float_as_uint(a) + 0x8000u;
  unsigned int ub = __float_as_uint(b) + 0x8000u;
  return __builtin_amdgcn_perm(ub, ua, 0x07060302u);
}
__device__ __forceinline__ u16 f2bf(float f) {
  return (u16)((__float_as_uint(f) + 0x8000u) >> 16);
}

union FragU { u32x4 q[2]; bf16x16 v; };

__device__ __forceinline__ v8f wmma_bf16(const FragU& a, const FragU& b, v8f c) {
  return __builtin_amdgcn_wmma_f32_16x16x32_bf16(false, a.v, false, b.v,
                                                 (short)0, c, false, false);
}

// ---------------------------------------------------------------------------
// prep: Hbase = z @ W1[:128]  (stored transposed, bf16)  [HID][N]
// ---------------------------------------------------------------------------
__global__ __launch_bounds__(256) void prep_hbase(const float* __restrict__ z,
                                                  const float* __restrict__ W1,
                                                  u16* __restrict__ HbT) {
  const int n = blockIdx.x;          // 384
  const int h = threadIdx.x;         // 256 == HID
  __shared__ float zs[IN_DIM_];
  if (h < IN_DIM_) zs[h] = z[(size_t)n * IN_DIM_ + h];
  __syncthreads();
  float acc = 0.f;
#pragma unroll 8
  for (int d = 0; d < IN_DIM_; ++d) acc += zs[d] * W1[(size_t)d * HID_ + h];
  HbT[(size_t)h * N_ + n] = f2bf(acc);
}

// prep: W2 transposed to bf16  [HID2][HID]
__global__ __launch_bounds__(256) void prep_w2t(const float* __restrict__ W2,
                                                u16* __restrict__ W2T) {
  const int idx = blockIdx.x * 256 + threadIdx.x;   // 32768
  const int h  = idx >> 7;
  const int k2 = idx & 127;
  W2T[(size_t)k2 * HID_ + h] = f2bf(W2[(size_t)h * HID2_ + k2]);
}

// prep: whole partials tensor f32 -> bf16 (bf16 copy fits in 192MB L2)
__global__ __launch_bounds__(256) void prep_pbf(const float* __restrict__ P,
                                                u16* __restrict__ Pbf) {
  const size_t idx = ((size_t)blockIdx.x * 256 + threadIdx.x) * 8;
  f32x4 f0 = *(const f32x4*)(P + idx);
  f32x4 f1 = *(const f32x4*)(P + idx + 4);
  u32x4 u = {pk2(f0[0], f0[1]), pk2(f0[2], f0[3]),
             pk2(f1[0], f1[1]), pk2(f1[2], f1[3])};
  *(u32x4*)(Pbf + idx) = u;
}

// ---------------------------------------------------------------------------
// Batched WMMA GEMM: C_b(M=384 x Nc) = A_b(M x K) @ B_b(K x Nc)
//   A: row-major (fp32 converted at commit time if AF32, else bf16)
//   B: column-major-by-output-column bf16: Bg[nc*ldb + k]
// Block tile 128x128, 8 waves (4M x 2N), wave tile 32x64, K-step 32.
// Double-buffered LDS; software-pipelined: global loads issued before the
// WMMA chain, LDS commit (and the loadcnt wait) after it.
// ---------------------------------------------------------------------------
template <int EPI, bool AF32>
__global__ __launch_bounds__(256)
void gemm_k(const void* __restrict__ Aall, long strideAbytes,
            const u16*  __restrict__ Ball, long strideBelem,
            int K, int lda, int ldb,
            void* __restrict__ Out,
            const float* __restrict__ partials,
            const float* __restrict__ w1last) {
  // padded rows: 40 u16 = 80 B (16B-aligned b128 frags, conflict-free)
  __shared__ u16 lgA[2][128 * 40];
  __shared__ u16 lgB[2][128 * 40];

  const int b     = blockIdx.z;
  const int mBase = blockIdx.x * 128;
  const int nBase = blockIdx.y * 128;
  const int tid   = threadIdx.x;
  const int wave  = tid >> 5;
  const int lane  = tid & 31;
  const int half  = lane >> 4;
  const int l16   = lane & 15;

  const char* Ab = (const char*)Aall + (size_t)b * (size_t)strideAbytes;
  const u16*  Bb = Ball + (size_t)b * (size_t)strideBelem;

  const int r  = tid >> 1;          // 0..127: tile row (A) / output column (B)
  const int c0 = (tid & 1) * 16;    // k-offset half

  // staging registers (live across the WMMA chain)
  f32x4 ra[4];        // AF32 path
  u32x4 rawA[2];      // bf16 path
  u32x4 rawB[2];

  auto issue_loads = [&](int k0) {
    if constexpr (AF32) {
      const float* ap = (const float*)Ab + (size_t)(mBase + r) * lda + k0 + c0;
      ra[0] = *(const f32x4*)(ap + 0);
      ra[1] = *(const f32x4*)(ap + 4);
      ra[2] = *(const f32x4*)(ap + 8);
      ra[3] = *(const f32x4*)(ap + 12);
      if (k0 + 32 < K) __builtin_prefetch(ap + 32, 0, 1);   // global_prefetch_b8
    } else {
      const u16* ap = (const u16*)Ab + (size_t)(mBase + r) * lda + k0 + c0;
      rawA[0] = *(const u32x4*)(ap);
      rawA[1] = *(const u32x4*)(ap + 8);
      if (k0 + 32 < K) __builtin_prefetch(ap + 32, 0, 1);
    }
    const u16* bp = Bb + (size_t)(nBase + r) * ldb + k0 + c0;
    rawB[0] = *(const u32x4*)(bp);
    rawB[1] = *(const u32x4*)(bp + 8);
  };

  auto commit_stage = [&](int buf) {
    u16* sA = &lgA[buf][0];
    u16* sB = &lgB[buf][0];
    if constexpr (AF32) {
      u32x4 u0 = {pk2(ra[0][0], ra[0][1]), pk2(ra[0][2], ra[0][3]),
                  pk2(ra[1][0], ra[1][1]), pk2(ra[1][2], ra[1][3])};
      u32x4 u1 = {pk2(ra[2][0], ra[2][1]), pk2(ra[2][2], ra[2][3]),
                  pk2(ra[3][0], ra[3][1]), pk2(ra[3][2], ra[3][3])};
      *(u32x4*)&sA[r * 40 + c0]     = u0;
      *(u32x4*)&sA[r * 40 + c0 + 8] = u1;
    } else {
      *(u32x4*)&sA[r * 40 + c0]     = rawA[0];
      *(u32x4*)&sA[r * 40 + c0 + 8] = rawA[1];
    }
    *(u32x4*)&sB[r * 40 + c0]     = rawB[0];
    *(u32x4*)&sB[r * 40 + c0 + 8] = rawB[1];
  };

  v8f acc[2][4];
#pragma unroll
  for (int i = 0; i < 2; ++i)
#pragma unroll
    for (int j = 0; j < 4; ++j)
      acc[i][j] = v8f{0.f, 0.f, 0.f, 0.f, 0.f, 0.f, 0.f, 0.f};

  const int nk = K >> 5;
  issue_loads(0);
  commit_stage(0);
  __syncthreads();

  const int mrow  = (wave & 3) * 32 + l16;
  const int ncol0 = (wave >> 2) * 64 + l16;

  for (int it = 0; it < nk; ++it) {
    const int cur = it & 1;
    const bool more = (it + 1 < nk);
    if (more) issue_loads((it + 1) << 5);   // global loads in flight over WMMAs

    const u16* sA = &lgA[cur][0];
    const u16* sB = &lgB[cur][0];
    FragU aF[2], bF[4];
#pragma unroll
    for (int wm = 0; wm < 2; ++wm) {
      const int off = (mrow + wm * 16) * 40 + half * 8;   // k = 8*half (+16)
      aF[wm].q[0] = *(const u32x4*)&sA[off];
      aF[wm].q[1] = *(const u32x4*)&sA[off + 16];
    }
#pragma unroll
    for (int wn = 0; wn < 4; ++wn) {
      const int off = (ncol0 + wn * 16) * 40 + half * 8;  // k = 8*half (+16)
      bF[wn].q[0] = *(const u32x4*)&sB[off];
      bF[wn].q[1] = *(const u32x4*)&sB[off + 16];
    }
#pragma unroll
    for (int wn = 0; wn < 4; ++wn) {
      acc[0][wn] = wmma_bf16(aF[0], bF[wn], acc[0][wn]);
      acc[1][wn] = wmma_bf16(aF[1], bF[wn], acc[1][wn]);
    }

    if (more) commit_stage(cur ^ 1);        // loadcnt wait lands here, post-WMMA
    __syncthreads();
  }

  // ---- epilogue ----
  if constexpr (EPI == 1) {
    u16* h2 = (u16*)Out + (size_t)b * (N_ * HID_);
#pragma unroll
    for (int wm = 0; wm < 2; ++wm) {
      const int m0 = mBase + (wave & 3) * 32 + wm * 16 + half * 8;
      float pc[8];
#pragma unroll
      for (int v = 0; v < 8; ++v)
        pc[v] = partials[(size_t)b * N_ * N_ + (size_t)(m0 + v) * N_ + b];
#pragma unroll
      for (int wn = 0; wn < 4; ++wn) {
        const int n_g = nBase + (wave >> 2) * 64 + wn * 16 + l16;
        const float wl = w1last[n_g];
#pragma unroll
        for (int v = 0; v < 8; ++v) {
          float val = acc[wm][wn][v] + pc[v] * wl;
          val = val > 0.f ? val : 0.f;                    // relu
          h2[(size_t)(m0 + v) * HID_ + n_g] = f2bf(val);
        }
      }
    }
  } else {
#pragma unroll
    for (int wm = 0; wm < 2; ++wm) {
#pragma unroll
      for (int wn = 0; wn < 4; ++wn) {
        const int n_g = nBase + (wave >> 2) * 64 + wn * 16 + l16;
        const int m0  = mBase + (wave & 3) * 32 + wm * 16 + half * 8;
        if constexpr (EPI == 2) {
          u16* h3T = (u16*)Out + (size_t)b * (HID2_ * N_);
          u32x4 u = {pk2(acc[wm][wn][0], acc[wm][wn][1]),
                     pk2(acc[wm][wn][2], acc[wm][wn][3]),
                     pk2(acc[wm][wn][4], acc[wm][wn][5]),
                     pk2(acc[wm][wn][6], acc[wm][wn][7])};
          *(u32x4*)&h3T[(size_t)n_g * N_ + m0] = u;
        } else {
          float* h4T = (float*)Out + (size_t)b * (HID2_ * N_);
          f32x4 s0 = {acc[wm][wn][0], acc[wm][wn][1], acc[wm][wn][2], acc[wm][wn][3]};
          f32x4 s1 = {acc[wm][wn][4], acc[wm][wn][5], acc[wm][wn][6], acc[wm][wn][7]};
          *(f32x4*)&h4T[(size_t)n_g * N_ + m0]     = s0;
          *(f32x4*)&h4T[(size_t)n_g * N_ + m0 + 4] = s1;
        }
      }
    }
  }
  (void)partials; (void)w1last;
}

// ---------------------------------------------------------------------------
// supplement: out[i,j] = x[i,j] + 0.5*sym(tril(<h4[i,i], h4[i,j]>))
// h4T layout [b][k][m] -> lane-coalesced reads along j
// ---------------------------------------------------------------------------
__global__ __launch_bounds__(256) void supp_kernel(const float* __restrict__ x,
                                                   const float* __restrict__ h4T,
                                                   float* __restrict__ out) {
  const int idx = blockIdx.x * 256 + threadIdx.x;   // 384*384
  const int i = idx / N_;
  const int j = idx % N_;
  if (j > i) return;                                 // lower-triangle owner
  const float* base = h4T + (size_t)i * (HID2_ * N_);
  float s = 0.f;
#pragma unroll 8
  for (int k = 0; k < HID2_; ++k)
    s += base[(size_t)k * N_ + i] * base[(size_t)k * N_ + j];
  s *= 0.5f;                                         // AUTOREGRESSIVE_SCALAR
  if (i == j) {
    out[(size_t)i * N_ + i] = x[(size_t)i * N_ + i] + 2.f * s;  // diag doubled
  } else {
    out[(size_t)i * N_ + j] = x[(size_t)i * N_ + j] + s;
    out[(size_t)j * N_ + i] = x[(size_t)j * N_ + i] + s;
  }
}

// ---------------------------------------------------------------------------
extern "C" void kernel_launch(void* const* d_in, const int* in_sizes, int n_in,
                              void* d_out, int out_size, void* d_ws, size_t ws_size,
                              hipStream_t stream) {
  (void)in_sizes; (void)n_in; (void)out_size;
  const float* z  = (const float*)d_in[0];   // [1,384,128]
  const float* x  = (const float*)d_in[1];   // [384,384]
  const float* P  = (const float*)d_in[2];   // [384,384,384]
  const float* W1 = (const float*)d_in[3];   // [129,256]
  const float* W2 = (const float*)d_in[4];   // [256,128]
  float* out = (float*)d_out;
  const float* w1last = W1 + (size_t)IN_DIM_ * HID_;

  char* ws = (char*)d_ws;
  u16* HbT = (u16*)(ws + 0);                         // [256][384] bf16  (192 KB)
  u16* W2T = (u16*)(ws + (size_t)(256 << 10));       // [128][256] bf16  (64 KB)

  const size_t SZ_PBF = (size_t)N_ * N_ * N_ * 2;    // 113,246,208
  const size_t SZ_H2  = (size_t)N_ * N_ * HID_ * 2;  //  75,497,472
  const size_t SZ_H3  = (size_t)N_ * N_ * HID2_ * 2; //  37,748,736
  const size_t OFF0   = (size_t)(1 << 20);

  prep_hbase<<<N_, 256, 0, stream>>>(z, W1, HbT);
  prep_w2t<<<HID2_, 256, 0, stream>>>(W2, W2T);

  if (ws_size >= OFF0 + SZ_PBF + SZ_H2 + SZ_H3) {
    // big-workspace path: pre-convert partials to bf16 (L2-resident)
    u16*  Pbf = (u16*)(ws + OFF0);
    u16*  h2  = (u16*)(ws + OFF0 + SZ_PBF);
    u16*  h3T = (u16*)(ws + OFF0 + SZ_PBF + SZ_H2);
    float* h4T = (float*)(ws + OFF0 + SZ_PBF);       // reuses h2 region (f32, same size)

    prep_pbf<<<(int)((size_t)N_ * N_ * N_ / (256 * 8)), 256, 0, stream>>>(P, Pbf);

    gemm_k<1, false><<<dim3(3, 2, N_), 256, 0, stream>>>(
        Pbf, (long)((size_t)N_ * N_ * 2), HbT, 0, N_, N_, N_, h2, P, w1last);
    gemm_k<2, false><<<dim3(3, 1, N_), 256, 0, stream>>>(
        h2, (long)((size_t)N_ * HID_ * 2), W2T, 0, HID_, HID_, HID_, h3T,
        nullptr, nullptr);
    gemm_k<3, false><<<dim3(3, 1, N_), 256, 0, stream>>>(
        Pbf, (long)((size_t)N_ * N_ * 2), h3T, (long)(HID2_ * N_), N_, N_, N_,
        h4T, nullptr, nullptr);
    supp_kernel<<<(N_ * N_) / 256, 256, 0, stream>>>(x, h4T, out);
  } else {
    // compact path: convert fp32 partials inline at commit time
    u16*  h2  = (u16*)(ws + OFF0);
    u16*  h3T = (u16*)(ws + OFF0 + SZ_H2);
    float* h4T = (float*)(ws + OFF0);                // reuses h2 region

    gemm_k<1, true><<<dim3(3, 2, N_), 256, 0, stream>>>(
        P, (long)((size_t)N_ * N_ * 4), HbT, 0, N_, N_, N_, h2, P, w1last);
    gemm_k<2, false><<<dim3(3, 1, N_), 256, 0, stream>>>(
        h2, (long)((size_t)N_ * HID_ * 2), W2T, 0, HID_, HID_, HID_, h3T,
        nullptr, nullptr);
    gemm_k<3, true><<<dim3(3, 1, N_), 256, 0, stream>>>(
        P, (long)((size_t)N_ * N_ * 4), h3T, (long)(HID2_ * N_), N_, N_, N_,
        h4T, nullptr, nullptr);
    supp_kernel<<<(N_ * N_) / 256, 256, 0, stream>>>(x, h4T, out);
  }
}
